// ModelMM_38585986188001
// MI455X (gfx1250) — matrix operand order repaired and checked by reading, not compiled, not run on hardware
//
#include <hip/hip_runtime.h>
#include <hip/hip_bf16.h>

// ---------------- problem constants ----------------
constexpr int Bb   = 128;          // batch
constexpr int Ll   = 50;           // seq len
constexpr int Dd   = 128;          // embed dim
constexpr int Nn   = Bb * (Ll + 1);    // 6528 columns
constexpr int NR   = Bb * Ll;          // 6400 rows
constexpr int NT   = Nn / 16;          // 408 column tiles
constexpr int NRT  = NR / 16;          // 400 row tiles
constexpr float NEGV  = -10000.0f;
constexpr float LOG2E = 1.4426950408889634f;
constexpr float LN2C  = 0.6931471805599453f;
constexpr float KILL2 = -1.0e9f;       // log2-domain "minus infinity" (exp2 -> 0)

// ---------------- types ----------------
typedef __attribute__((ext_vector_type(16))) __bf16 v16bf;
typedef __attribute__((ext_vector_type(8)))  float  v8f;

union Frag16 { uint4 q[2]; v16bf v; };

static __device__ __forceinline__ unsigned short f2bf(float f) {
    unsigned u = __float_as_uint(f);
    unsigned r = u + 0x7FFFu + ((u >> 16) & 1u);   // round-to-nearest-even
    return (unsigned short)(r >> 16);
}
static __device__ __forceinline__ float bf2f(unsigned short h) {
    return __uint_as_float(((unsigned)h) << 16);
}

// ---- prep: split f32 -> bf16 hi/lo, interleaved [row][Dd hi | Dd lo] ----
__global__ void cvt_split_kernel(const float* __restrict__ src,
                                 unsigned short* __restrict__ dst,
                                 int n, float scale) {
    int i = blockIdx.x * blockDim.x + threadIdx.x;
    if (i < n) {
        int row = i / Dd;
        int k   = i - row * Dd;
        float v = src[i] * scale;
        unsigned short h = f2bf(v);
        dst[(size_t)row * (2 * Dd) + k]      = h;
        dst[(size_t)row * (2 * Dd) + Dd + k] = f2bf(v - bf2f(h));
    }
}

// ---- prep: per-column bias (-log2 pop), invalid column -> -1e9 ----
__global__ void col_kernel(const int* __restrict__ ids,
                           const int* __restrict__ log_mask,
                           const float* __restrict__ pop,
                           float* __restrict__ colAdd2) {
    int k = blockIdx.x * blockDim.x + threadIdx.x;
    if (k < Nn) {
        int i = k / (Ll + 1);
        int t = k - i * (Ll + 1);
        bool valid = (t == Ll) || (log_mask[i * Ll + t] != 0);
        colAdd2[k] = valid ? -log2f(pop[ids[k]]) : KILL2;
    }
}

// ---- prep: membership table member[i][k] ----
__global__ void member_kernel(const int* __restrict__ ids,
                              unsigned char* __restrict__ member) {
    int idx = blockIdx.x * blockDim.x + threadIdx.x;
    if (idx < Bb * Nn) {
        int i = idx / Nn;
        int k = idx - i * Nn;
        int v = ids[k];
        const int* lst = ids + i * (Ll + 1);
        int m = 0;
        for (int l = 0; l < Ll + 1; ++l) m |= (lst[l] == v);
        member[idx] = (unsigned char)m;
    }
}

// ---- prep: exact-f32 target logit per row ----
__global__ void tgt_kernel(const float* __restrict__ prec,
                           const float* __restrict__ embs,
                           const int* __restrict__ ids,
                           const int* __restrict__ log_mask,
                           const float* __restrict__ pop,
                           float* __restrict__ tgtLogit) {
    int r = blockIdx.x * blockDim.x + threadIdx.x;
    if (r < NR) {
        int i = r / Ll;
        int j = r - i * Ll;
        int tgt = i * (Ll + 1) + j + 1;
        bool valid = (j + 1 == Ll) || (log_mask[i * Ll + j + 1] != 0);
        const float4* a = (const float4*)(prec + (size_t)r * Dd);
        const float4* b = (const float4*)(embs + (size_t)tgt * Dd);
        float s = 0.f;
#pragma unroll
        for (int q = 0; q < Dd / 4; ++q) {
            float4 x = a[q], y = b[q];
            s += x.x * y.x + x.y * y.y + x.z * y.z + x.w * y.w;
        }
        tgtLogit[r] = valid ? (s - logf(pop[ids[tgt]])) : NEGV;
    }
}

// ---- main fused GEMM + masked sum-of-exp2 sweep ----
// All member columns are killed; each row's own target contribution is added
// back exactly in the epilogue (the target column is always a member column).
__launch_bounds__(256)
__global__ void sweep_kernel(const unsigned short* __restrict__ A2,   // [row][hi|lo]
                             const unsigned short* __restrict__ B2,   // [col][hi|lo]
                             const float* __restrict__ colAdd2,
                             const unsigned char* __restrict__ member,
                             const float* __restrict__ tgtLogit,
                             const int* __restrict__ log_mask,
                             float* __restrict__ rowLoss) {
    __shared__ float part[8 * 32 * 8];   // [wave][lane][r]
    __shared__ float comb[16 * 8];       // [row-in-tile][wave]

    const int tid  = threadIdx.x;
    const int w    = tid >> 5;
    const int lane = tid & 31;
    const int half = lane >> 4;
    const int lq   = lane & 15;
    const int tileRow = blockIdx.x * 16;

    // --- load A fragments once (ISA 16-bit A 16x32 layout; hi/lo interleaved) ---
    Frag16 ahi[4], alo[4];
    {
        const unsigned short* pA = A2 + (size_t)(tileRow + lq) * (2 * Dd);
#pragma unroll
        for (int c = 0; c < 4; ++c) {
            const int base = c * 32 + half * 8;
            ahi[c].q[0] = *(const uint4*)(pA + base);
            ahi[c].q[1] = *(const uint4*)(pA + base + 16);
            alo[c].q[0] = *(const uint4*)(pA + Dd + base);
            alo[c].q[1] = *(const uint4*)(pA + Dd + base + 16);
        }
    }

    // --- per-lane row constants (C layout: M = r + 8*half) ---
    // This lane's 8 rows are consecutive -> at most 2 distinct users.
    const int rbase = tileRow + 8 * half;
    const int u0 = rbase / Ll;
    const int u1 = (rbase + 7) / Ll;
    const unsigned char* mrow0 = member + (size_t)u0 * Nn;
    const unsigned char* mrow1 = member + (size_t)u1 * Nn;

    bool sel1[8];
#pragma unroll
    for (int r = 0; r < 8; ++r) sel1[r] = ((rbase + r) / Ll != u0);

    float s[8];
#pragma unroll
    for (int r = 0; r < 8; ++r) s[r] = 0.f;

    // --- sweep column tiles: wave w handles tiles w, w+8, ... ---
    for (int t = w; t < NT; t += 8) {
        const int col = t * 16 + lq;
        v8f acc = {0.f, 0.f, 0.f, 0.f, 0.f, 0.f, 0.f, 0.f};
        const unsigned short* pB = B2 + (size_t)col * (2 * Dd);
#pragma unroll
        for (int c = 0; c < 4; ++c) {
            const int kb = c * 32 + half * 8;   // ISA 16-bit B 32x16 layout
            Frag16 bh, bl;
            bh.q[0] = *(const uint4*)(pB + kb);
            bh.q[1] = *(const uint4*)(pB + kb + 16);
            bl.q[0] = *(const uint4*)(pB + Dd + kb);
            bl.q[1] = *(const uint4*)(pB + Dd + kb + 16);
            acc = __builtin_amdgcn_wmma_f32_16x16x32_bf16(false, ahi[c].v, false, bh.v,
                                                          (short)0, acc, false, false);
            acc = __builtin_amdgcn_wmma_f32_16x16x32_bf16(false, ahi[c].v, false, bl.v,
                                                          (short)0, acc, false, false);
            acc = __builtin_amdgcn_wmma_f32_16x16x32_bf16(false, alo[c].v, false, bh.v,
                                                          (short)0, acc, false, false);
        }
        const float cAdd = colAdd2[col];             // includes column-invalid kill
        const float a0 = (mrow0[col] != 0) ? KILL2 : cAdd;
        const float a1 = (mrow1[col] != 0) ? KILL2 : cAdd;
#pragma unroll
        for (int r = 0; r < 8; ++r) {
            const float add = sel1[r] ? a1 : a0;     // loop-invariant lane mask
            s[r] += exp2f(acc[r] + add);             // fixed-shift sum of exp2
        }
    }

#pragma unroll
    for (int r = 0; r < 8; ++r)
        part[((w << 5) + lane) * 8 + r] = s[r];
    __syncthreads();

    // --- stage 1: sum 16 lanes of one wave for one row ---
    if (tid < 128) {
        const int rr = tid >> 3, ww = tid & 7;
        const int l0 = (rr & 8) ? 16 : 0;
        const int r  = rr & 7;
        float S = 0.f;
        for (int c2 = 0; c2 < 16; ++c2)
            S += part[((ww << 5) + l0 + c2) * 8 + r];
        comb[rr * 8 + ww] = S;
    }
    __syncthreads();

    // --- stage 2: sum 8 waves; add back exact target term; emit per-row loss ---
    if (tid < 16) {
        float S = 0.f;
        for (int ww = 0; ww < 8; ++ww) S += comb[tid * 8 + ww];
        const int row = tileRow + tid;
        const float tl = tgtLogit[row];
        S += exp2f(tl * LOG2E);                      // re-insert target column
        const float lse = log2f(S) * LN2C;           // back to natural log
        const bool valid = log_mask[row] != 0;
        rowLoss[row] = valid ? (lse - tl) : 0.f;
    }
}

// ---- deterministic final reduction ----
__global__ void finalize_kernel(const float* __restrict__ rowLoss,
                                const int* __restrict__ log_mask,
                                float* __restrict__ out) {
    __shared__ float sl[256];
    __shared__ int   sc[256];
    const int tid = threadIdx.x;
    float s = 0.f; int c = 0;
    for (int i = tid; i < NR; i += 256) {
        s += rowLoss[i];
        c += (log_mask[i] != 0);
    }
    sl[tid] = s; sc[tid] = c;
    __syncthreads();
    for (int o = 128; o > 0; o >>= 1) {
        if (tid < o) { sl[tid] += sl[tid + o]; sc[tid] += sc[tid + o]; }
        __syncthreads();
    }
    if (tid == 0) out[0] = sl[0] / (float)sc[0];
}

// ---- workspace layout (bytes, 256-aligned) ----
constexpr size_t ALN(size_t x) { return (x + 255) & ~(size_t)255; }
constexpr size_t OFF_A2   = 0;
constexpr size_t OFF_B2   = OFF_A2 + ALN((size_t)NR * 2 * Dd * 2);
constexpr size_t OFF_CADD = OFF_B2 + ALN((size_t)Nn * 2 * Dd * 2);
constexpr size_t OFF_MEMB = OFF_CADD + ALN((size_t)Nn * 4);
constexpr size_t OFF_TGT  = OFF_MEMB + ALN((size_t)Bb * Nn);
constexpr size_t OFF_RLOSS = OFF_TGT + ALN((size_t)NR * 4);

extern "C" void kernel_launch(void* const* d_in, const int* in_sizes, int n_in,
                              void* d_out, int out_size, void* d_ws, size_t ws_size,
                              hipStream_t stream) {
    (void)in_sizes; (void)n_in; (void)out_size; (void)ws_size;
    const int*   ids      = (const int*)d_in[0];
    const int*   log_mask = (const int*)d_in[1];
    const float* prec     = (const float*)d_in[2];
    const float* embs     = (const float*)d_in[3];
    const float* pop      = (const float*)d_in[4];
    float* out = (float*)d_out;

    char* ws = (char*)d_ws;
    unsigned short* A2   = (unsigned short*)(ws + OFF_A2);
    unsigned short* B2   = (unsigned short*)(ws + OFF_B2);
    float*          cAdd = (float*)(ws + OFF_CADD);
    unsigned char*  memb = (unsigned char*)(ws + OFF_MEMB);
    float*          tgtL = (float*)(ws + OFF_TGT);
    float*          rLoss = (float*)(ws + OFF_RLOSS);

    const int nA = NR * Dd;   // 819200
    const int nB = Nn * Dd;   // 835584

    cvt_split_kernel<<<(nA + 255) / 256, 256, 0, stream>>>(prec, A2, nA, LOG2E);
    cvt_split_kernel<<<(nB + 255) / 256, 256, 0, stream>>>(embs, B2, nB, 1.0f);
    col_kernel<<<(Nn + 255) / 256, 256, 0, stream>>>(ids, log_mask, pop, cAdd);
    member_kernel<<<(Bb * Nn + 255) / 256, 256, 0, stream>>>(ids, memb);
    tgt_kernel<<<(NR + 255) / 256, 256, 0, stream>>>(prec, embs, ids, log_mask, pop, tgtL);
    sweep_kernel<<<NRT, 256, 0, stream>>>(A2, B2, cAdd, memb, tgtL, log_mask, rLoss);
    finalize_kernel<<<1, 256, 0, stream>>>(rLoss, log_mask, out);
}
